// DynamicEdgeDiscriminator_14809047236960
// MI455X (gfx1250) — compile-verified, matrix operand order checked by reading, not run on hardware
//
#include <hip/hip_runtime.h>

#define BB   8
#define NN   2048
#define KNbr 10

typedef __attribute__((ext_vector_type(16))) __bf16 v16bf;
typedef __attribute__((ext_vector_type(8)))  float  v8f;

// ---------------------------------------------------------------------------
// WMMA operand fragment layout (CDNA5 ISA 7.12.2, 16-bit operands, wave32):
//   lane = (n&15) + 16*((k>>3)&1), elem = (k&7) + 8*(k>>4)
// Tiles are stored fragment-ready (512 bf16 = 1KB per 16x32 tile), so a lane's
// 16 elements are 32 contiguous bytes -> fragment load = 2x *_load_b128.
// Producers (prep/combine/packW) write this layout directly to GLOBAL memory,
// so consumers need no LDS staging and no barriers around the WMMA loops.
// ---------------------------------------------------------------------------
__device__ __forceinline__ int fragIdx(int k, int n) {
  return (((n & 15) + (((k >> 3) & 1) << 4)) << 4) + (k & 7) + ((k >> 4) << 3);
}

__device__ __forceinline__ v16bf load_frag(const __bf16* fragBase, int lane) {
  const uint4* p = (const uint4*)fragBase + lane * 2;
  union { uint4 u[2]; v16bf v; } u;
  u.u[0] = p[0];
  u.u[1] = p[1];
  return u.v;
}

// ---------------------------------------------------------------------------
// Prep: x (B,N,3) fp32 -> fragment-ready bf16 tensor Xf (kstep=1, zero-padded
// to K=32) + per-point squared norms.
// ---------------------------------------------------------------------------
__global__ void prep_kernel(const float* __restrict__ x, __bf16* __restrict__ Xf,
                            float* __restrict__ sq) {
  int i = blockIdx.x * 256 + threadIdx.x;
  if (i >= BB * NN) return;
  int b = i / NN, n = i - b * NN;
  float s = 0.f;
  __bf16 vals[3];
#pragma unroll
  for (int c = 0; c < 3; ++c) {
    vals[c] = (__bf16)x[i * 3 + c];
    float fv = (float)vals[c];
    s += fv * fv;
  }
  sq[i] = s;
  __bf16* base = Xf + ((size_t)b * 128 + (n >> 4)) * 512;
#pragma unroll
  for (int k = 0; k < 32; ++k)
    base[fragIdx(k, n & 15)] = (k < 3) ? vals[k] : (__bf16)0.f;
}

// ---------------------------------------------------------------------------
// Weight pre-pack: W (2*fin, fout) fp32 -> bf16 fragment-ready
// Wp[ks][cotile][512] for the fused [W1 | W2] operand (zero pad k >= fin).
// ---------------------------------------------------------------------------
__global__ void packW_kernel(const float* __restrict__ W, __bf16* __restrict__ Wp,
                             int fin, int fout) {
  const int twofout = 2 * fout;
  const int ksteps  = (fin + 31) >> 5;
  const int total   = ksteps * 32 * twofout;
  int e = blockIdx.x * 256 + threadIdx.x;
  if (e >= total) return;
  int co = e % twofout;
  int kf = e / twofout;
  float wv = 0.f;
  if (kf < fin)
    wv = (co < fout) ? W[(size_t)kf * fout + co]
                     : W[(size_t)(fin + kf) * fout + (co - fout)];
  Wp[((size_t)(kf >> 5) * (twofout >> 4) + (co >> 4)) * 512 +
     fragIdx(kf & 31, co & 15)] = (__bf16)wv;
}

// ---------------------------------------------------------------------------
// KNN: block = 16 query rows. Gram via WMMA with operands loaded directly
// from the fragment-ready global tensor; fused top-10 per row through LDS.
// KSTEPS = fin/32 rounded up; for KSTEPS<=8 the query panel stays in VGPRs.
// ---------------------------------------------------------------------------
template <int KSTEPS>
__global__ __launch_bounds__(128)
void knn_kernel(const __bf16* __restrict__ Xf, const float* __restrict__ sq,
                int* __restrict__ idx) {
  __shared__ float d2t[16][260];
  __shared__ float sqc[256];
  __shared__ float sqr[16];

  const int tid  = threadIdx.x;
  const int lane = tid & 31;
  const int wid  = tid >> 5;
  const int rt   = blockIdx.x;           // row tile 0..127
  const int b    = blockIdx.y;
  const __bf16* Xb = Xf + (size_t)b * KSTEPS * 128 * 512;

  constexpr bool AREG = (KSTEPS <= 8);
  v16bf af[KSTEPS];
  if constexpr (AREG) {
#pragma unroll
    for (int ks = 0; ks < KSTEPS; ++ks)
      af[ks] = load_frag(Xb + ((size_t)ks * 128 + rt) * 512, lane);
  }
  if (tid < 16) sqr[tid] = sq[b * NN + rt * 16 + tid];

  float best[KNbr];
  int   bidx[KNbr];
#pragma unroll
  for (int k = 0; k < KNbr; ++k) { best[k] = 3.4e38f; bidx[k] = 0; }

  for (int chunk = 0; chunk < NN / 256; ++chunk) {
    const int cbase = chunk * 256;
    sqc[tid]       = sq[b * NN + cbase + tid];
    sqc[tid + 128] = sq[b * NN + cbase + tid + 128];

    v8f acc[4];
#pragma unroll
    for (int j = 0; j < 4; ++j)
#pragma unroll
      for (int v = 0; v < 8; ++v) acc[j][v] = 0.f;

    if constexpr (AREG) {
#pragma unroll
      for (int j = 0; j < 4; ++j) {
        const int tile = chunk * 16 + wid * 4 + j;
#pragma unroll
        for (int ks = 0; ks < KSTEPS; ++ks) {
          v16bf bf = load_frag(Xb + ((size_t)ks * 128 + tile) * 512, lane);
          acc[j] = __builtin_amdgcn_wmma_f32_16x16x32_bf16(
              false, af[ks], false, bf, (short)0, acc[j], false, false);
        }
      }
    } else {
      for (int ks = 0; ks < KSTEPS; ++ks) {
        v16bf a = load_frag(Xb + ((size_t)ks * 128 + rt) * 512, lane);
#pragma unroll
        for (int j = 0; j < 4; ++j) {
          const int tile = chunk * 16 + wid * 4 + j;
          v16bf bf = load_frag(Xb + ((size_t)ks * 128 + tile) * 512, lane);
          acc[j] = __builtin_amdgcn_wmma_f32_16x16x32_bf16(
              false, a, false, bf, (short)0, acc[j], false, false);
        }
      }
    }
    __syncthreads();                      // sqc visible; prev selection done

    const int rowadd = (lane >> 4) << 3;
#pragma unroll
    for (int j = 0; j < 4; ++j) {
      int col = (wid * 4 + j) * 16 + (lane & 15);
#pragma unroll
      for (int v = 0; v < 8; ++v) {
        int r = v + rowadd;
        d2t[r][col] = sqr[r] + sqc[col] - 2.f * acc[j][v];
      }
    }
    __syncthreads();

    if (tid < 16) {
      for (int c = 0; c < 256; ++c) {
        float d = d2t[tid][c];
        if (d < best[KNbr - 1]) {
          float nv = d; int ni = cbase + c;
#pragma unroll
          for (int k = 0; k < KNbr; ++k) {
            bool  sw = nv < best[k];
            float tb = best[k]; int ti = bidx[k];
            best[k] = sw ? nv : tb; bidx[k] = sw ? ni : ti;
            nv = sw ? tb : nv;      ni = sw ? ti : ni;
          }
        }
      }
    }
    __syncthreads();
  }

  if (tid < 16) {
    int* op = idx + ((size_t)b * NN + rt * 16 + tid) * KNbr;
#pragma unroll
    for (int k = 0; k < KNbr; ++k) op[k] = bidx[k];
  }
}

// ---------------------------------------------------------------------------
// Dual GEMM, LDS-free / barrier-free: one wave = 32 rows x 64 cols macro tile.
// A from fragment-ready Xf, B from pre-packed Wp -> per kstep: 6 fragment
// loads (12x global_load_b128) + 8 WMMAs.
// ---------------------------------------------------------------------------
__global__ __launch_bounds__(256)
void gemm_dual_kernel(const __bf16* __restrict__ Xf, const __bf16* __restrict__ Wp,
                      const float* __restrict__ bias, float* __restrict__ C,
                      int ksteps, int fout) {
  const int tid  = threadIdx.x;
  const int lane = tid & 31;
  const int wid  = tid >> 5;
  const int twofout = 2 * fout;
  const int ntY  = twofout >> 6;          // 64-col macro tiles
  const int t    = blockIdx.x * 8 + wid;  // global wave id
  if (t >= BB * 64 * ntY) return;         // wave-uniform; no barriers here
  const int ty = t % ntY;
  const int tx = (t / ntY) & 63;
  const int b  = t / (ntY * 64);
  const int rowtile = tx * 2;             // in 16-row tiles
  const int coltile = ty * 4;             // in 16-col tiles

  v8f acc[2][4];
#pragma unroll
  for (int rt = 0; rt < 2; ++rt)
#pragma unroll
    for (int j = 0; j < 4; ++j)
#pragma unroll
      for (int v = 0; v < 8; ++v) acc[rt][j][v] = 0.f;

  const __bf16* Xb = Xf + (size_t)b * ksteps * 128 * 512;
  for (int ks = 0; ks < ksteps; ++ks) {
    v16bf a0 = load_frag(Xb + ((size_t)ks * 128 + rowtile) * 512, lane);
    v16bf a1 = load_frag(Xb + ((size_t)ks * 128 + rowtile + 1) * 512, lane);
    const __bf16* Wks = Wp + ((size_t)ks * (twofout >> 4) + coltile) * 512;
#pragma unroll
    for (int j = 0; j < 4; ++j) {
      v16bf bf = load_frag(Wks + j * 512, lane);
      acc[0][j] = __builtin_amdgcn_wmma_f32_16x16x32_bf16(
          false, a0, false, bf, (short)0, acc[0][j], false, false);
      acc[1][j] = __builtin_amdgcn_wmma_f32_16x16x32_bf16(
          false, a1, false, bf, (short)0, acc[1][j], false, false);
    }
  }

  const int rowadd = (lane >> 4) << 3;
#pragma unroll
  for (int j = 0; j < 4; ++j) {
    const int colg = (coltile + j) * 16 + (lane & 15);
    const float bia = (colg < fout) ? bias[colg] : 0.f;
#pragma unroll
    for (int rt = 0; rt < 2; ++rt) {
      float* Cp = C + ((size_t)b * NN + (rowtile + rt) * 16) * twofout + colg;
#pragma unroll
      for (int v = 0; v < 8; ++v)
        Cp[(size_t)(v + rowadd) * twofout] = acc[rt][j][v] + bia;
    }
  }
}

// ---------------------------------------------------------------------------
// Combine: out[i,f] = max_k leaky( hc[i,f] + P[j_k,f] - P[i,f] ).
// Writes fp32 hF (final pooling), fragment-ready bf16 Xf (next layer), norms.
// ---------------------------------------------------------------------------
__global__ __launch_bounds__(256)
void edge_combine_kernel(const float* __restrict__ CP, const int* __restrict__ idx,
                         float* __restrict__ hF, __bf16* __restrict__ XfOut,
                         float* __restrict__ sq, int fout) {
  __shared__ float red[256];
  const int    tid = threadIdx.x;
  const size_t pt  = blockIdx.x;         // b*NN + n
  const int    b   = (int)(pt / NN);
  const int    n   = (int)(pt - (size_t)b * NN);
  const int twofout = 2 * fout;
  const int ksn     = fout >> 5;          // ksteps of the produced features

  int nb[KNbr];
#pragma unroll
  for (int k = 0; k < KNbr; ++k) nb[k] = idx[pt * KNbr + k];
#pragma unroll
  for (int k = 0; k < KNbr; ++k)
    __builtin_prefetch(CP + ((size_t)b * NN + nb[k]) * twofout + fout, 0, 0);

  const float* rowC = CP + pt * twofout;
  float ssum = 0.f;
  for (int f = tid; f < fout; f += 256) {
    float hc = rowC[f];
    float pv = rowC[fout + f];
    float m  = -3.4e38f;
#pragma unroll
    for (int k = 0; k < KNbr; ++k) {
      const float* pj = CP + ((size_t)b * NN + nb[k]) * twofout + fout;
      float h = hc + pj[f] - pv;
      h = h > 0.f ? h : 0.2f * h;
      m = fmaxf(m, h);
    }
    __bf16 mb = (__bf16)m;
    hF[pt * fout + f] = m;
    XfOut[((size_t)(b * ksn + (f >> 5)) * 128 + (n >> 4)) * 512 +
          fragIdx(f & 31, n & 15)] = mb;
    float mf = (float)mb;
    ssum += mf * mf;
  }
  red[tid] = ssum;
  __syncthreads();
  for (int s = 128; s > 0; s >>= 1) {
    if (tid < s) red[tid] += red[tid + s];
    __syncthreads();
  }
  if (tid == 0) sq[pt] = red[0];
}

// ---------------------------------------------------------------------------
// Global max pool over N per feature.
// ---------------------------------------------------------------------------
__global__ __launch_bounds__(256)
void gmax_kernel(const float* __restrict__ hF, float* __restrict__ g) {
  int f = blockIdx.x * 256 + threadIdx.x;   // 0..1023
  int b = blockIdx.y;
  float m = -3.4e38f;
  const float* p = hF + (size_t)b * NN * 1024 + f;
  for (int n = 0; n < NN; ++n) m = fmaxf(m, p[(size_t)n * 1024]);
  g[b * 1024 + f] = m;
}

// ---------------------------------------------------------------------------
// MLP head: 1024 -> 128 -> 64 -> 1 (no activations), one block per batch.
// ---------------------------------------------------------------------------
__global__ __launch_bounds__(256)
void mlp_kernel(const float* __restrict__ g,
                const float* __restrict__ m0w, const float* __restrict__ m0b,
                const float* __restrict__ m1w, const float* __restrict__ m1b,
                const float* __restrict__ m2w, const float* __restrict__ m2b,
                float* __restrict__ out) {
  __shared__ float gg[1024];
  __shared__ float h1[128];
  __shared__ float h2[64];
  const int tid = threadIdx.x;
  const int b   = blockIdx.x;
#pragma unroll
  for (int i = 0; i < 4; ++i) gg[tid + i * 256] = g[b * 1024 + tid + i * 256];
  __syncthreads();
  if (tid < 128) {
    float s = m0b[tid];
    for (int k = 0; k < 1024; ++k) s += gg[k] * m0w[k * 128 + tid];
    h1[tid] = s;
  }
  __syncthreads();
  if (tid < 64) {
    float s = m1b[tid];
    for (int k = 0; k < 128; ++k) s += h1[k] * m1w[k * 64 + tid];
    h2[tid] = s;
  }
  __syncthreads();
  if (tid == 0) {
    float s = m2b[0];
    for (int k = 0; k < 64; ++k) s += h2[k] * m2w[k];
    out[b] = s;
  }
}

// ---------------------------------------------------------------------------
extern "C" void kernel_launch(void* const* d_in, const int* in_sizes, int n_in,
                              void* d_out, int out_size, void* d_ws, size_t ws_size,
                              hipStream_t stream) {
  const float* x = (const float*)d_in[0];
  const float* w[5]; const float* bias[5];
  for (int l = 0; l < 5; ++l) {
    w[l]    = (const float*)d_in[1 + 2 * l];
    bias[l] = (const float*)d_in[2 + 2 * l];
  }
  const float* m0w = (const float*)d_in[11];
  const float* m0b = (const float*)d_in[12];
  const float* m1w = (const float*)d_in[13];
  const float* m1b = (const float*)d_in[14];
  const float* m2w = (const float*)d_in[15];
  const float* m2b = (const float*)d_in[16];
  float* out = (float*)d_out;

  char* ws = (char*)d_ws;
  size_t o = 0;
  auto wsalloc = [&](size_t bytes) -> void* {
    void* p = ws + o;
    o += (bytes + 255) & ~(size_t)255;
    return p;
  };
  // fragment-ready feature tensors (max ksteps written = 1024/32 = 32)
  __bf16* XfA = (__bf16*)wsalloc((size_t)BB * 32 * 128 * 512 * 2);
  __bf16* XfB = (__bf16*)wsalloc((size_t)BB * 32 * 128 * 512 * 2);
  __bf16* Wp  = (__bf16*)wsalloc((size_t)16 * 32 * 2048 * 2 * 2);
  float*  hF  = (float*)wsalloc((size_t)BB * NN * 1024 * 4);
  float*  CP  = (float*)wsalloc((size_t)BB * NN * 2048 * 4);
  int*    idx = (int*)wsalloc((size_t)BB * NN * KNbr * 4);
  float*  sq  = (float*)wsalloc((size_t)BB * NN * 4);
  float*  g   = (float*)wsalloc((size_t)BB * 1024 * 4);

  const int FIN[5]  = {3, 64, 128, 256, 512};
  const int FOUT[5] = {64, 128, 256, 512, 1024};

  prep_kernel<<<(BB * NN + 255) / 256, 256, 0, stream>>>(x, XfA, sq);

  __bf16* cur = XfA;
  __bf16* nxt = XfB;
  for (int l = 0; l < 5; ++l) {
    const int fin = FIN[l], fout = FOUT[l];
    const int ksteps = (fin + 31) >> 5;

    switch (ksteps) {
      case 1:  knn_kernel<1><<<dim3(128, BB), 128, 0, stream>>>(cur, sq, idx); break;
      case 2:  knn_kernel<2><<<dim3(128, BB), 128, 0, stream>>>(cur, sq, idx); break;
      case 4:  knn_kernel<4><<<dim3(128, BB), 128, 0, stream>>>(cur, sq, idx); break;
      case 8:  knn_kernel<8><<<dim3(128, BB), 128, 0, stream>>>(cur, sq, idx); break;
      default: knn_kernel<16><<<dim3(128, BB), 128, 0, stream>>>(cur, sq, idx); break;
    }

    const int packTotal = ksteps * 32 * 2 * fout;
    packW_kernel<<<(packTotal + 255) / 256, 256, 0, stream>>>(w[l], Wp, fin, fout);

    const int ntY = (2 * fout) >> 6;
    const int waves = BB * 64 * ntY;
    gemm_dual_kernel<<<(waves + 7) / 8, 256, 0, stream>>>(cur, Wp, bias[l], CP,
                                                          ksteps, fout);

    edge_combine_kernel<<<BB * NN, 256, 0, stream>>>(CP, idx, hF, nxt, sq, fout);
    __bf16* t = cur; cur = nxt; nxt = t;
  }
  gmax_kernel<<<dim3(1024 / 256, BB), 256, 0, stream>>>(hF, g);
  mlp_kernel<<<BB, 256, 0, stream>>>(g, m0w, m0b, m1w, m1b, m2w, m2b, out);

  (void)in_sizes; (void)n_in; (void)out_size; (void)ws_size;
}
